// RecommendationModel_38774964748344
// MI455X (gfx1250) — compile-verified, matrix operand order checked by reading, not run on hardware
//
#include <hip/hip_runtime.h>

typedef __attribute__((ext_vector_type(16))) __bf16 v16bf;
typedef __attribute__((ext_vector_type(8)))  __bf16 v8bf;
typedef __attribute__((ext_vector_type(8)))  float  v8f;

__device__ inline v16bf pack16(v8bf lo, v8bf hi) {
  v16bf b;
#pragma unroll
  for (int i = 0; i < 8; ++i) { b[i] = lo[i]; b[8 + i] = hi[i]; }
  return b;
}

// ---------------------------------------------------------------------------
// Y[M,C] = X[M,K] @ W[K,C] + bias, bf16 WMMA, f32 accumulate.
// Wt is transposed+converted: Wt[c*K + k] = bf16(W[k*C + c]) -> B fragments
// are contiguous 16B runs. Each wave owns a 16x64 output tile with 4
// independent accumulators (no WMMA->WMMA RAW chain inside a k-step, A
// fragment reused 4x). Block = 4 waves -> 16 x 256 cols per block.
// A layout (16-bit A 16x32): lane m = lane&15, kh = lane>>4 picks K-half;
//   elem e<8 : K = k0 + kh*8 + e ; elem e>=8 : K = k0 + 16 + kh*8 + (e-8).
// D layout: elem r -> row = tile*16 + r + 8*kh, col = subtile base + (lane&15).
// ---------------------------------------------------------------------------
template<int K>
__global__ __launch_bounds__(128)
void gemm_bf16_wmma(const float* __restrict__ X, int ldX,
                    const __bf16* __restrict__ Wt,
                    const float* __restrict__ bias,
                    float* __restrict__ Y, int ldY, int C, int M) {
  const int lane = threadIdx.x & 31;
  const int wave = threadIdx.x >> 5;
  const int m    = lane & 15;
  const int kh   = lane >> 4;                    // 0 or 1 (K-half)
  const int col0 = blockIdx.y * 256 + wave * 64; // wave's 64-col slab
  if (col0 >= C) return;                         // wave-uniform exit
  long row = (long)blockIdx.x * 16 + m;
  if (row >= M) row = M - 1;                     // clamp (M=50000 is 16-divisible)

  const float* xrow = X + row * (long)ldX;
  const __bf16* w[4];
#pragma unroll
  for (int j = 0; j < 4; ++j) w[j] = Wt + (long)(col0 + j * 16 + m) * K;

  v8f acc[4] = {{}, {}, {}, {}};
#pragma unroll
  for (int k0 = 0; k0 < K; k0 += 32) {
    // ---- A fragment: two contiguous runs of 8 f32, convert to bf16 ----
    const float* ap = xrow + k0 + kh * 8;
    float4 A0 = *(const float4*)(ap);
    float4 A1 = *(const float4*)(ap + 4);
    float4 A2 = *(const float4*)(ap + 16);
    float4 A3 = *(const float4*)(ap + 20);
    v16bf a;
    a[0]=(__bf16)A0.x;  a[1]=(__bf16)A0.y;  a[2]=(__bf16)A0.z;  a[3]=(__bf16)A0.w;
    a[4]=(__bf16)A1.x;  a[5]=(__bf16)A1.y;  a[6]=(__bf16)A1.z;  a[7]=(__bf16)A1.w;
    a[8]=(__bf16)A2.x;  a[9]=(__bf16)A2.y;  a[10]=(__bf16)A2.z; a[11]=(__bf16)A2.w;
    a[12]=(__bf16)A3.x; a[13]=(__bf16)A3.y; a[14]=(__bf16)A3.z; a[15]=(__bf16)A3.w;
    // ---- 4 B fragments, 4 independent WMMAs reusing A ----
#pragma unroll
    for (int j = 0; j < 4; ++j) {
      v8bf B0 = *(const v8bf*)(w[j] + k0 + kh * 8);
      v8bf B1 = *(const v8bf*)(w[j] + k0 + 16 + kh * 8);
      v16bf b = pack16(B0, B1);
      acc[j] = __builtin_amdgcn_wmma_f32_16x16x32_bf16(false, a, false, b,
                                                       (short)0, acc[j], false, false);
    }
  }

#pragma unroll
  for (int j = 0; j < 4; ++j) {
    const int n = col0 + j * 16 + m;
    const float bv = bias[n];
#pragma unroll
    for (int r = 0; r < 8; ++r) {
      const long orow = (long)blockIdx.x * 16 + r + 8 * kh;
      if (orow < M) Y[orow * (long)ldY + n] = acc[j][r] + bv;
    }
  }
}

// Wt[c*K + k] = bf16(W[k*C + c])
__global__ void transpose_to_bf16(const float* __restrict__ W, __bf16* __restrict__ Wt,
                                  int K, int C) {
  int t = blockIdx.x * blockDim.x + threadIdx.x;
  if (t >= K * C) return;
  int c = t / K, k = t - c * K;
  Wt[t] = (__bf16)W[k * C + c];
}

__global__ void count_kernel(const int* __restrict__ dst, float* __restrict__ cnt, int E) {
  int e = blockIdx.x * blockDim.x + threadIdx.x;
  if (e < E) atomicAdd(&cnt[dst[e]], 1.0f);
}

// agg[dst*C + f] += V[src*ldV + f]; feature-parallel float4 per thread.
template<int C>
__global__ void scatter_add_kernel(const int* __restrict__ src, const int* __restrict__ dst,
                                   const float* __restrict__ V, int ldV,
                                   float* __restrict__ agg, int E) {
  const int G = C / 4;
  long t = (long)blockIdx.x * blockDim.x + threadIdx.x;
  long e = t / G;
  if (e >= E) return;
  int f = (int)(t - e * G) * 4;
  long s = src[e], d = dst[e];
  const float4 v = *(const float4*)(V + s * (long)ldV + f);
  float* a = agg + d * (long)C + f;
  atomicAdd(a + 0, v.x);
  atomicAdd(a + 1, v.y);
  atomicAdd(a + 2, v.z);
  atomicAdd(a + 3, v.w);
}

// out[i*ldOut+c] = [relu]( agg[i*C+c]/max(cnt[i],1) + lin[i*ldLin+c] )
// (out may alias lin when ldOut==ldLin)
__global__ void combine_kernel(const float* __restrict__ agg, const float* __restrict__ cnt,
                               const float* __restrict__ lin, int ldLin,
                               float* __restrict__ out, int ldOut,
                               int N, int C, int relu) {
  long t = (long)blockIdx.x * blockDim.x + threadIdx.x;
  if (t >= (long)N * C) return;
  int i = (int)(t / C);
  int c = (int)(t - (long)i * C);
  float v = agg[t] / fmaxf(cnt[i], 1.0f) + lin[(long)i * ldLin + c];
  v = relu ? fmaxf(v, 0.0f) : v;
  out[(long)i * ldOut + c] = v;
}

// sigmoid( [u_emb | i_emb] . head_W + head_b ) ; wave-per-node, wave32 reduce.
__global__ __launch_bounds__(128)
void head_kernel(const float* __restrict__ ue, const float* __restrict__ ie,
                 const float* __restrict__ hW, const float* __restrict__ hb,
                 float* __restrict__ out, int N) {
  int lane = threadIdx.x & 31;
  int node = blockIdx.x * 4 + (threadIdx.x >> 5);
  if (node >= N) return;
  long b = (long)node * 64;
  float s = ue[b + lane]      * hW[lane]
          + ue[b + 32 + lane] * hW[32 + lane]
          + ie[b + lane]      * hW[64 + lane]
          + ie[b + 32 + lane] * hW[96 + lane];
#pragma unroll
  for (int o = 16; o > 0; o >>= 1) s += __shfl_xor(s, o, 32);
  if (lane == 0) out[node] = 1.0f / (1.0f + __expf(-(s + hb[0])));
}

// ---------------------------------------------------------------------------
// One SAGE encoder: fused [Wn|Wr] GEMMs, atomic mean-aggregation, fused
// mean+root+bias(+ReLU) combine. Layer-2 projects to 64 feats BEFORE
// aggregating (linearity of segment-sum) -> 4x less edge traffic.
// ---------------------------------------------------------------------------
static void run_encoder(const float* x, const int* src, const int* dst, int N, int E,
                        const __bf16* wtcat1, const float* biascat1,   // [512,256], [512]
                        const __bf16* wtcat2, const float* biascat2,   // [128,256], [128]
                        float* ycat1,  // [N,512]: cols 0..255 = x@Wn, 256..511 = x@Wr+b
                        float* agg1,   // [N,256]
                        float* ycat2,  // [N,128]: cols 0..63 = h@Wn, 64..127 = h@Wr+b
                        float* agg2,   // [N,64]
                        float* cnt,    // [N]
                        float* emb,    // [N,64]
                        hipStream_t stream) {
  const int HID = 256, EMB = 64;
  dim3 blk(128);
  // layer 1 fused GEMM: [N,256] @ [256,512]
  gemm_bf16_wmma<256><<<dim3((N + 15) / 16, 2), blk, 0, stream>>>(
      x, 256, wtcat1, biascat1, ycat1, 512, 512, N);
  hipMemsetAsync(agg1, 0, (size_t)N * HID * sizeof(float), stream);
  hipMemsetAsync(cnt, 0, (size_t)N * sizeof(float), stream);
  count_kernel<<<(E + 255) / 256, 256, 0, stream>>>(dst, cnt, E);
  {
    long total = (long)E * (HID / 4);
    scatter_add_kernel<256><<<(int)((total + 255) / 256), 256, 0, stream>>>(
        src, dst, ycat1 /*xn half*/, 512, agg1, E);
  }
  {
    long total = (long)N * HID;  // h = relu(agg/cnt + x@Wr + b), in place over root half
    combine_kernel<<<(int)((total + 255) / 256), 256, 0, stream>>>(
        agg1, cnt, ycat1 + 256, 512, ycat1 + 256, 512, N, HID, 1);
  }
  // layer 2 fused GEMM: [N,256] @ [256,128] (X = hidden state, strided rows)
  gemm_bf16_wmma<256><<<dim3((N + 15) / 16, 1), blk, 0, stream>>>(
      ycat1 + 256, 512, wtcat2, biascat2, ycat2, 128, 128, N);
  hipMemsetAsync(agg2, 0, (size_t)N * EMB * sizeof(float), stream);
  {
    long total = (long)E * (EMB / 4);
    scatter_add_kernel<64><<<(int)((total + 255) / 256), 256, 0, stream>>>(
        src, dst, ycat2 /*hn half*/, 128, agg2, E);
  }
  {
    long total = (long)N * EMB;
    combine_kernel<<<(int)((total + 255) / 256), 256, 0, stream>>>(
        agg2, cnt, ycat2 + 64, 128, emb, 64, N, EMB, 0);
  }
}

extern "C" void kernel_launch(void* const* d_in, const int* in_sizes, int n_in,
                              void* d_out, int out_size, void* d_ws, size_t ws_size,
                              hipStream_t stream) {
  (void)n_in; (void)out_size; (void)ws_size;
  const int IN = 256, HID = 256, EMB = 64;
  const int N = in_sizes[0] / IN;
  const int E = in_sizes[2] / 2;

  const float* user_x = (const float*)d_in[0];
  const float* item_x = (const float*)d_in[1];
  const int* u_edge   = (const int*)d_in[2];   // [2,E]: row0 = src, row1 = dst
  const int* i_edge   = (const int*)d_in[3];
  const float* u1_Wn  = (const float*)d_in[4];
  const float* u1_Wr  = (const float*)d_in[5];
  const float* u1_b   = (const float*)d_in[6];
  const float* u2_Wn  = (const float*)d_in[7];
  const float* u2_Wr  = (const float*)d_in[8];
  const float* u2_b   = (const float*)d_in[9];
  const float* i1_Wn  = (const float*)d_in[10];
  const float* i1_Wr  = (const float*)d_in[11];
  const float* i1_b   = (const float*)d_in[12];
  const float* i2_Wn  = (const float*)d_in[13];
  const float* i2_Wr  = (const float*)d_in[14];
  const float* i2_b   = (const float*)d_in[15];
  const float* head_W = (const float*)d_in[16];
  const float* head_b = (const float*)d_in[17];
  float* out = (float*)d_out;

  // ---- workspace carve-up (512B aligned slabs) ----
  char* ws = (char*)d_ws;
  size_t off = 0;
  auto alloc = [&](size_t bytes) -> void* {
    void* p = ws + off;
    off += (bytes + 511) & ~(size_t)511;
    return p;
  };
  // fused transposed weights: layer1 [512 cols][256 k], layer2 [128 cols][256 k]
  __bf16* wtc1_u = (__bf16*)alloc((size_t)512 * IN * sizeof(__bf16));
  __bf16* wtc2_u = (__bf16*)alloc((size_t)128 * HID * sizeof(__bf16));
  __bf16* wtc1_i = (__bf16*)alloc((size_t)512 * IN * sizeof(__bf16));
  __bf16* wtc2_i = (__bf16*)alloc((size_t)128 * HID * sizeof(__bf16));
  float* bc1_u = (float*)alloc(512 * sizeof(float));
  float* bc2_u = (float*)alloc(128 * sizeof(float));
  float* bc1_i = (float*)alloc(512 * sizeof(float));
  float* bc2_i = (float*)alloc(128 * sizeof(float));
  float* ycat1 = (float*)alloc((size_t)N * 512 * sizeof(float));
  float* agg1  = (float*)alloc((size_t)N * HID * sizeof(float));
  float* ycat2 = (float*)alloc((size_t)N * 128 * sizeof(float));
  float* agg2  = (float*)alloc((size_t)N * EMB * sizeof(float));
  float* cnt   = (float*)alloc((size_t)N * sizeof(float));
  float* emb_u = (float*)alloc((size_t)N * EMB * sizeof(float));
  float* emb_i = (float*)alloc((size_t)N * EMB * sizeof(float));

  // ---- weights: transpose + convert to bf16 into fused [Wn|Wr] slabs ----
  {
    const int t1 = IN * HID, t2 = HID * EMB;
    transpose_to_bf16<<<(t1 + 255) / 256, 256, 0, stream>>>(u1_Wn, wtc1_u, IN, HID);
    transpose_to_bf16<<<(t1 + 255) / 256, 256, 0, stream>>>(u1_Wr, wtc1_u + (size_t)HID * IN, IN, HID);
    transpose_to_bf16<<<(t2 + 255) / 256, 256, 0, stream>>>(u2_Wn, wtc2_u, HID, EMB);
    transpose_to_bf16<<<(t2 + 255) / 256, 256, 0, stream>>>(u2_Wr, wtc2_u + (size_t)EMB * HID, HID, EMB);
    transpose_to_bf16<<<(t1 + 255) / 256, 256, 0, stream>>>(i1_Wn, wtc1_i, IN, HID);
    transpose_to_bf16<<<(t1 + 255) / 256, 256, 0, stream>>>(i1_Wr, wtc1_i + (size_t)HID * IN, IN, HID);
    transpose_to_bf16<<<(t2 + 255) / 256, 256, 0, stream>>>(i2_Wn, wtc2_i, HID, EMB);
    transpose_to_bf16<<<(t2 + 255) / 256, 256, 0, stream>>>(i2_Wr, wtc2_i + (size_t)EMB * HID, HID, EMB);
  }
  // ---- fused bias vectors: [0 .. 0 | b] ----
  hipMemsetAsync(bc1_u, 0, HID * sizeof(float), stream);
  hipMemcpyAsync(bc1_u + HID, u1_b, HID * sizeof(float), hipMemcpyDeviceToDevice, stream);
  hipMemsetAsync(bc2_u, 0, EMB * sizeof(float), stream);
  hipMemcpyAsync(bc2_u + EMB, u2_b, EMB * sizeof(float), hipMemcpyDeviceToDevice, stream);
  hipMemsetAsync(bc1_i, 0, HID * sizeof(float), stream);
  hipMemcpyAsync(bc1_i + HID, i1_b, HID * sizeof(float), hipMemcpyDeviceToDevice, stream);
  hipMemsetAsync(bc2_i, 0, EMB * sizeof(float), stream);
  hipMemcpyAsync(bc2_i + EMB, i2_b, EMB * sizeof(float), hipMemcpyDeviceToDevice, stream);

  run_encoder(user_x, u_edge, u_edge + E, N, E,
              wtc1_u, bc1_u, wtc2_u, bc2_u,
              ycat1, agg1, ycat2, agg2, cnt, emb_u, stream);
  run_encoder(item_x, i_edge, i_edge + E, N, E,
              wtc1_i, bc1_i, wtc2_i, bc2_i,
              ycat1, agg1, ycat2, agg2, cnt, emb_i, stream);

  head_kernel<<<(N + 3) / 4, 128, 0, stream>>>(emb_u, emb_i, head_W, head_b, out, N);
}
